// DrugGVPModel_27865747817104
// MI455X (gfx1250) — compile-verified, matrix operand order checked by reading, not run on hardware
//
#include <hip/hip_runtime.h>
#include <hip/hip_bf16.h>

// GVP-GNN forward for MI455X (gfx1250, wave32, WMMA).
// All dense matmuls use v_wmma_f32_16x16x32_bf16. Weights are pre-packed into
// WMMA *fragment order* (32-lane x 16-bf16 contiguous per 32x16 block) so each
// B fragment is one 32B contiguous load; A fragments are two ds_load_b128.
// Workspace use: ~241 MB (fp32 state + bf16-packed weights).

typedef __attribute__((ext_vector_type(16))) __bf16 v16bf;
typedef __attribute__((ext_vector_type(16))) unsigned short v16u;
typedef __attribute__((ext_vector_type(8)))  unsigned short v8u;
typedef __attribute__((ext_vector_type(8)))  float v8f;

#define DEVF static __device__ __forceinline__

DEVF unsigned short f2bf(float f) {
  unsigned int u = __builtin_bit_cast(unsigned int, f);
  u += 0x7FFFu + ((u >> 16) & 1u);           // round-to-nearest-even
  return (unsigned short)(u >> 16);
}
DEVF float sigm(float x) { return 1.f / (1.f + __expf(-x)); }

// A fragment: 16x32 bf16 tile from row-major LDS (ld elements, 16B-aligned
// rows). Per ISA layout lane's 16 values are rows m=lane&15, K runs
// [kb..kb+7] and [kb+16..kb+23] with kb = (lane&16)?8:0 -> two b128 loads.
DEVF v16bf frag_a(const unsigned short* p, int ld, int k0, int lane) {
  const int m = lane & 15;
  const int kb = (lane & 16) ? 8 : 0;
  const unsigned short* row = p + m * ld + k0 + kb;
  const v8u lo = *(const v8u*)(row);
  const v8u hi = *(const v8u*)(row + 16);
  v16u t = __builtin_shufflevector(lo, hi, 0, 1, 2, 3, 4, 5, 6, 7,
                                   8, 9, 10, 11, 12, 13, 14, 15);
  return __builtin_bit_cast(v16bf, t);
}
// B fragment from swizzled weights: lane's 16 bf16 are contiguous.
DEVF v16bf frag_b(const unsigned short* p) {
  const v16u t = *(const v16u*)p;
  return __builtin_bit_cast(v16bf, t);
}
DEVF v8f wmma_bf(v16bf a, v16bf b, v8f c) {
  return __builtin_amdgcn_wmma_f32_16x16x32_bf16(false, a, false, b, (short)0, c,
                                                 false, false);
}
// 16x16 C tile = A(16xKp LDS) @ B(KpxN swizzled weights), K stepped by 32.
// Swizzled layout: block (kb,nb) at ((nb*(Kp/32))+kb)*512 + lane*16.
DEVF v8f gemm_tile(const unsigned short* A, int lda, const unsigned short* Bsw,
                   int n0, int Kp, int lane) {
  v8f acc = {0.f, 0.f, 0.f, 0.f, 0.f, 0.f, 0.f, 0.f};
  const int kblocks = Kp >> 5;
  const unsigned short* bt = Bsw + (size_t)(n0 >> 4) * kblocks * 512 + lane * 16;
  for (int kb = 0; kb < kblocks; ++kb)
    acc = wmma_bf(frag_a(A, lda, kb * 32, lane), frag_b(bt + kb * 512), acc);
  return acc;
}
DEVF float wred(float x) {
#pragma unroll
  for (int o = 16; o > 0; o >>= 1) x += __shfl_xor(x, o, 32);
  return x;
}

// --------- weight packing: f32 KxN -> bf16 fragment-swizzled KpxNp ----------
__global__ void pack_w(const float* __restrict__ W, unsigned short* __restrict__ O,
                       int K, int N, int Kp, int Np) {
  const int i = blockIdx.x * blockDim.x + threadIdx.x;
  if (i >= Kp * Np) return;
  const int e = i & 15, lane = (i >> 4) & 31, blk = i >> 9;
  const int kblocks = Kp >> 5;
  const int nb = blk / kblocks, kb = blk - nb * kblocks;
  const int n = nb * 16 + (lane & 15);
  const int r = e >> 1;
  const int k = kb * 32 + ((r & 4) << 2) + ((lane & 16) ? 8 : 0) +
                ((r & 3) << 1) + (e & 1);
  O[i] = (k < K && n < N) ? f2bf(W[k * N + n]) : (unsigned short)0;
}

__global__ void deg_kernel(const int* __restrict__ ei, float* __restrict__ deg,
                           int E) {
  int e = blockIdx.x * blockDim.x + threadIdx.x;
  if (e < E) atomicAdd(&deg[ei[E + e]], 1.0f);
}

// ---------------- node embedding: LN + GVP(66,1 -> 128,64), no act ----------
__global__ __launch_bounds__(256) void node_embed(
    const float* __restrict__ ns, const float* __restrict__ nv,
    const float* __restrict__ g, const float* __restrict__ b,
    const float* __restrict__ wh,            // (1,64) f32
    const unsigned short* __restrict__ wsP,  // 160x128 swizzled
    const float* __restrict__ wsb,           // 128
    const unsigned short* __restrict__ wvP,  // 64x64 swizzled
    float* __restrict__ s, float* __restrict__ v) {
  __shared__ alignas(16) unsigned short catA[16][160];
  __shared__ alignas(16) unsigned short vhb[3][16][64];
  __shared__ float vnm[16][3];
  const int tid = threadIdx.x, lane = tid & 31, wave = tid >> 5;
  const int n0 = blockIdx.x * 16;
  for (int i = tid; i < 16 * 160; i += 256) (&catA[0][0])[i] = 0;
  __syncthreads();
  if (tid < 16) {
    const int node = n0 + tid;
    float mu = 0.f;
    for (int k = 0; k < 66; ++k) mu += ns[node * 66 + k];
    mu *= (1.f / 66.f);
    float var = 0.f;
    for (int k = 0; k < 66; ++k) { float d = ns[node * 66 + k] - mu; var += d * d; }
    var *= (1.f / 66.f);
    const float inv = rsqrtf(var + 1e-5f);
    for (int k = 0; k < 66; ++k)
      catA[tid][k] = f2bf((ns[node * 66 + k] - mu) * inv * g[k] + b[k]);
    const float x = nv[node * 3], y = nv[node * 3 + 1], z = nv[node * 3 + 2];
    const float invn = rsqrtf(fmaxf(x * x + y * y + z * z, 1e-8f)); // vi==1
    vnm[tid][0] = x * invn; vnm[tid][1] = y * invn; vnm[tid][2] = z * invn;
  }
  __syncthreads();
  for (int i = tid; i < 16 * 64; i += 256) {       // vh = outer(v_hat, wh)
    const int m = i >> 6, h = i & 63;
    const float w0 = wh[h];
    const float a0 = vnm[m][0] * w0, a1 = vnm[m][1] * w0, a2 = vnm[m][2] * w0;
    vhb[0][m][h] = f2bf(a0); vhb[1][m][h] = f2bf(a1); vhb[2][m][h] = f2bf(a2);
    catA[m][66 + h] = f2bf(sqrtf(fmaxf(a0 * a0 + a1 * a1 + a2 * a2, 1e-8f)));
  }
  __syncthreads();
  {  // s' = [s_ln, vn] @ ws + b   (K=160, N=128)
    v8f acc = gemm_tile(&catA[0][0], 160, wsP, wave * 16, 160, lane);
    const int col = wave * 16 + (lane & 15), mb = (lane & 16) ? 8 : 0;
#pragma unroll
    for (int r = 0; r < 8; ++r)
      s[(n0 + mb + r) * 128 + col] = acc[r] + wsb[col];
  }
  for (int t = wave; t < 4; t += 8) {  // vout = vh @ wv  (K=64, N=64)
    v8f a0 = gemm_tile(&vhb[0][0][0], 64, wvP, t * 16, 64, lane);
    v8f a1 = gemm_tile(&vhb[1][0][0], 64, wvP, t * 16, 64, lane);
    v8f a2 = gemm_tile(&vhb[2][0][0], 64, wvP, t * 16, 64, lane);
    const int o = t * 16 + (lane & 15), mb = (lane & 16) ? 8 : 0;
#pragma unroll
    for (int r = 0; r < 8; ++r) {
      const int node = n0 + mb + r;
      v[node * 192 + o * 3 + 0] = a0[r];
      v[node * 192 + o * 3 + 1] = a1[r];
      v[node * 192 + o * 3 + 2] = a2[r];
    }
  }
}

// ---------------- edge embedding: LN + GVP(16,1 -> 32,1), scalar ------------
__global__ void edge_embed(const float* __restrict__ es_in,
                           const float* __restrict__ ev_in,
                           const float* __restrict__ g, const float* __restrict__ b,
                           const float* __restrict__ wh, const float* __restrict__ ws,
                           const float* __restrict__ wsb, const float* __restrict__ wv,
                           float* __restrict__ es, float* __restrict__ ev, int E) {
  const int e = blockIdx.x * blockDim.x + threadIdx.x;
  if (e >= E) return;
  float sv[16];
  float mu = 0.f;
  for (int k = 0; k < 16; ++k) { sv[k] = es_in[e * 16 + k]; mu += sv[k]; }
  mu *= (1.f / 16.f);
  float var = 0.f;
  for (int k = 0; k < 16; ++k) { float d = sv[k] - mu; var += d * d; }
  var *= (1.f / 16.f);
  const float inv = rsqrtf(var + 1e-5f);
  for (int k = 0; k < 16; ++k) sv[k] = (sv[k] - mu) * inv * g[k] + b[k];
  const float x = ev_in[e * 3], y = ev_in[e * 3 + 1], z = ev_in[e * 3 + 2];
  const float invn = rsqrtf(fmaxf(x * x + y * y + z * z, 1e-8f));
  const float w0 = wh[0];
  const float vx = x * invn * w0, vy = y * invn * w0, vz = z * invn * w0;
  const float vn = sqrtf(fmaxf(vx * vx + vy * vy + vz * vz, 1e-8f));
  for (int o = 0; o < 32; ++o) {
    float acc = wsb[o];
    for (int k = 0; k < 16; ++k) acc += sv[k] * ws[k * 32 + o];
    acc += vn * ws[16 * 32 + o];
    es[e * 32 + o] = acc;
  }
  const float w1 = wv[0];
  ev[e * 3 + 0] = vx * w1; ev[e * 3 + 1] = vy * w1; ev[e * 3 + 2] = vz * w1;
}

// ---------------- fused message path: 3 chained GVPs per 16-edge tile -------
struct MsgW {
  const unsigned short *wh0, *ws0, *wv0;   // Kp x Np: 160x144, 448x128, 160x64
  const unsigned short *wh1, *ws1, *wv1;   // 64x64, 192x128, 64x64
  const unsigned short *wh2, *ws2, *wv2;
  const float *b0, *b1, *b2;
};

__global__ __launch_bounds__(256) void msg_fused(
    const float* __restrict__ s, const float* __restrict__ v,
    const float* __restrict__ es, const float* __restrict__ ev,
    const int* __restrict__ ei, MsgW w, float* __restrict__ ds,
    float* __restrict__ dv, int E) {
  __shared__ alignas(16) unsigned short sA[16][448];     // [s_j|es|s_i|vn0]
  __shared__ alignas(16) unsigned short vbf[3][16][160]; // v concat -> vout
  __shared__ alignas(16) unsigned short vhb[3][16][160]; // vh per GVP
  __shared__ alignas(16) unsigned short sB[16][192];     // [s_k | vn_k]
  __shared__ int srcn[16], dstn[16];
  const int tid = threadIdx.x, lane = tid & 31, wave = tid >> 5;
  const int e0 = blockIdx.x * 16;
  for (int i = tid; i < 16 * 448; i += 256) (&sA[0][0])[i] = 0;
  for (int i = tid; i < 3 * 16 * 160; i += 256) {
    (&vbf[0][0][0])[i] = 0; (&vhb[0][0][0])[i] = 0;
  }
  if (tid < 16) { srcn[tid] = ei[e0 + tid]; dstn[tid] = ei[E + e0 + tid]; }
  __syncthreads();
  // gather: sA cols [0:128)=s_j [128:160)=es [160:288)=s_i ; v concat (129)
  for (int i = tid; i < 16 * 128; i += 256) {
    const int m = i >> 7, k = i & 127;
    sA[m][k]       = f2bf(s[srcn[m] * 128 + k]);
    sA[m][160 + k] = f2bf(s[dstn[m] * 128 + k]);
  }
  for (int i = tid; i < 16 * 32; i += 256) {
    const int m = i >> 5, k = i & 31;
    sA[m][128 + k] = f2bf(es[(e0 + m) * 32 + k]);
  }
  for (int i = tid; i < 16 * 64; i += 256) {
    const int m = i >> 6, o = i & 63;
#pragma unroll
    for (int c = 0; c < 3; ++c) {
      vbf[c][m][o]      = f2bf(v[srcn[m] * 192 + o * 3 + c]);
      vbf[c][m][65 + o] = f2bf(v[dstn[m] * 192 + o * 3 + c]);
    }
  }
  if (tid < 16)
#pragma unroll
    for (int c = 0; c < 3; ++c) vbf[c][tid][64] = f2bf(ev[(e0 + tid) * 3 + c]);
  __syncthreads();
  // GVP0 vh: (16x160)@(160x144) per coord; vn0 -> sA[:,288..]
  for (int t = wave; t < 9; t += 8) {
    v8f a0 = gemm_tile(&vbf[0][0][0], 160, w.wh0, t * 16, 160, lane);
    v8f a1 = gemm_tile(&vbf[1][0][0], 160, w.wh0, t * 16, 160, lane);
    v8f a2 = gemm_tile(&vbf[2][0][0], 160, w.wh0, t * 16, 160, lane);
    v8f n2 = a0 * a0 + a1 * a1 + a2 * a2;
    const int col = t * 16 + (lane & 15), mb = (lane & 16) ? 8 : 0;
#pragma unroll
    for (int r = 0; r < 8; ++r) {
      const int m = mb + r;
      vhb[0][m][col] = f2bf(a0[r]);
      vhb[1][m][col] = f2bf(a1[r]);
      vhb[2][m][col] = f2bf(a2[r]);
      sA[m][288 + col] = f2bf(sqrtf(fmaxf(n2[r], 1e-8f)));
    }
  }
  __syncthreads();
  // GVP0 s: (16x448)@(448x128)+b relu -> sB[:,0:128); v: wv0 + sigmoid gate
  {
    v8f aS = gemm_tile(&sA[0][0], 448, w.ws0, wave * 16, 448, lane);
    const int col = wave * 16 + (lane & 15), mb = (lane & 16) ? 8 : 0;
#pragma unroll
    for (int r = 0; r < 8; ++r)
      sB[mb + r][col] = f2bf(fmaxf(aS[r] + w.b0[col], 0.f));
  }
  for (int t = wave; t < 4; t += 8) {
    v8f a0 = gemm_tile(&vhb[0][0][0], 160, w.wv0, t * 16, 160, lane);
    v8f a1 = gemm_tile(&vhb[1][0][0], 160, w.wv0, t * 16, 160, lane);
    v8f a2 = gemm_tile(&vhb[2][0][0], 160, w.wv0, t * 16, 160, lane);
    v8f n2 = a0 * a0 + a1 * a1 + a2 * a2;
    const int col = t * 16 + (lane & 15), mb = (lane & 16) ? 8 : 0;
#pragma unroll
    for (int r = 0; r < 8; ++r) {
      const int m = mb + r;
      const float gt = sigm(sqrtf(fmaxf(n2[r], 1e-8f)));
      vbf[0][m][col] = f2bf(a0[r] * gt);
      vbf[1][m][col] = f2bf(a1[r] * gt);
      vbf[2][m][col] = f2bf(a2[r] * gt);
    }
  }
  __syncthreads();
  // GVP1 vh (K=64,N=64); vn1 -> sB[:,128..]
  for (int t = wave; t < 4; t += 8) {
    v8f a0 = gemm_tile(&vbf[0][0][0], 160, w.wh1, t * 16, 64, lane);
    v8f a1 = gemm_tile(&vbf[1][0][0], 160, w.wh1, t * 16, 64, lane);
    v8f a2 = gemm_tile(&vbf[2][0][0], 160, w.wh1, t * 16, 64, lane);
    v8f n2 = a0 * a0 + a1 * a1 + a2 * a2;
    const int col = t * 16 + (lane & 15), mb = (lane & 16) ? 8 : 0;
#pragma unroll
    for (int r = 0; r < 8; ++r) {
      const int m = mb + r;
      vhb[0][m][col] = f2bf(a0[r]); vhb[1][m][col] = f2bf(a1[r]);
      vhb[2][m][col] = f2bf(a2[r]);
      sB[m][128 + col] = f2bf(sqrtf(fmaxf(n2[r], 1e-8f)));
    }
  }
  __syncthreads();
  // GVP1 s/v in regs, then overwrite sB / vbf
  {
    v8f aS = gemm_tile(&sB[0][0], 192, w.ws1, wave * 16, 192, lane);
    v8f a0 = {0, 0, 0, 0, 0, 0, 0, 0}, a1 = a0, a2 = a0;
    const bool hv = wave < 4;
    if (hv) {
      a0 = gemm_tile(&vhb[0][0][0], 160, w.wv1, wave * 16, 64, lane);
      a1 = gemm_tile(&vhb[1][0][0], 160, w.wv1, wave * 16, 64, lane);
      a2 = gemm_tile(&vhb[2][0][0], 160, w.wv1, wave * 16, 64, lane);
    }
    __syncthreads();
    const int col = wave * 16 + (lane & 15), mb = (lane & 16) ? 8 : 0;
#pragma unroll
    for (int r = 0; r < 8; ++r)
      sB[mb + r][col] = f2bf(fmaxf(aS[r] + w.b1[col], 0.f));
    if (hv) {
      v8f n2 = a0 * a0 + a1 * a1 + a2 * a2;
#pragma unroll
      for (int r = 0; r < 8; ++r) {
        const int m = mb + r;
        const float gt = sigm(sqrtf(fmaxf(n2[r], 1e-8f)));
        vbf[0][m][col] = f2bf(a0[r] * gt);
        vbf[1][m][col] = f2bf(a1[r] * gt);
        vbf[2][m][col] = f2bf(a2[r] * gt);
      }
    }
  }
  __syncthreads();
  // GVP2 vh; vn2 -> sB[:,128..]
  for (int t = wave; t < 4; t += 8) {
    v8f a0 = gemm_tile(&vbf[0][0][0], 160, w.wh2, t * 16, 64, lane);
    v8f a1 = gemm_tile(&vbf[1][0][0], 160, w.wh2, t * 16, 64, lane);
    v8f a2 = gemm_tile(&vbf[2][0][0], 160, w.wh2, t * 16, 64, lane);
    v8f n2 = a0 * a0 + a1 * a1 + a2 * a2;
    const int col = t * 16 + (lane & 15), mb = (lane & 16) ? 8 : 0;
#pragma unroll
    for (int r = 0; r < 8; ++r) {
      const int m = mb + r;
      vhb[0][m][col] = f2bf(a0[r]); vhb[1][m][col] = f2bf(a1[r]);
      vhb[2][m][col] = f2bf(a2[r]);
      sB[m][128 + col] = f2bf(sqrtf(fmaxf(n2[r], 1e-8f)));
    }
  }
  __syncthreads();
  // GVP2 outputs (no act): atomic scatter to target nodes (mean later)
  {
    v8f aS = gemm_tile(&sB[0][0], 192, w.ws2, wave * 16, 192, lane);
    const int col = wave * 16 + (lane & 15), mb = (lane & 16) ? 8 : 0;
#pragma unroll
    for (int r = 0; r < 8; ++r)
      atomicAdd(&ds[dstn[mb + r] * 128 + col], aS[r] + w.b2[col]);
  }
  for (int t = wave; t < 4; t += 8) {
    v8f a0 = gemm_tile(&vhb[0][0][0], 160, w.wv2, t * 16, 64, lane);
    v8f a1 = gemm_tile(&vhb[1][0][0], 160, w.wv2, t * 16, 64, lane);
    v8f a2 = gemm_tile(&vhb[2][0][0], 160, w.wv2, t * 16, 64, lane);
    const int o = t * 16 + (lane & 15), mb = (lane & 16) ? 8 : 0;
#pragma unroll
    for (int r = 0; r < 8; ++r) {
      const int node = dstn[mb + r];
      atomicAdd(&dv[node * 192 + o * 3 + 0], a0[r]);
      atomicAdd(&dv[node * 192 + o * 3 + 1], a1[r]);
      atomicAdd(&dv[node * 192 + o * 3 + 2], a2[r]);
    }
  }
}

// ---------------- fused feed-forward: 2 chained GVPs per 16-node tile -------
struct FFW {
  const unsigned short *wh0, *ws0, *wv0;   // 64x128, 256x512, 128x128
  const unsigned short *wh1, *ws1, *wv1;   // 128x128, 640x128, 128x64
  const float *b0, *b1;
};

__global__ __launch_bounds__(256) void ff_fused(
    const float* __restrict__ s, const float* __restrict__ v, FFW w,
    float* __restrict__ fs, float* __restrict__ fv) {
  __shared__ alignas(16) unsigned short catA[16][640];
  __shared__ alignas(16) unsigned short vbf[3][16][128];
  __shared__ alignas(16) unsigned short vhb[3][16][128];
  const int tid = threadIdx.x, lane = tid & 31, wave = tid >> 5;
  const int n0 = blockIdx.x * 16;
  for (int i = tid; i < 16 * 128; i += 256) {
    const int m = i >> 7, k = i & 127;
    catA[m][k] = f2bf(s[(n0 + m) * 128 + k]);
  }
  for (int i = tid; i < 16 * 64; i += 256) {
    const int m = i >> 6, o = i & 63;
#pragma unroll
    for (int c = 0; c < 3; ++c) vbf[c][m][o] = f2bf(v[(n0 + m) * 192 + o * 3 + c]);
  }
  __syncthreads();
  // GVP0 vh (K=64, N=128); vn0 -> catA[:,128..256)
  {
    const int t = wave;
    v8f a0 = gemm_tile(&vbf[0][0][0], 128, w.wh0, t * 16, 64, lane);
    v8f a1 = gemm_tile(&vbf[1][0][0], 128, w.wh0, t * 16, 64, lane);
    v8f a2 = gemm_tile(&vbf[2][0][0], 128, w.wh0, t * 16, 64, lane);
    v8f n2 = a0 * a0 + a1 * a1 + a2 * a2;
    const int col = t * 16 + (lane & 15), mb = (lane & 16) ? 8 : 0;
#pragma unroll
    for (int r = 0; r < 8; ++r) {
      const int m = mb + r;
      vhb[0][m][col] = f2bf(a0[r]); vhb[1][m][col] = f2bf(a1[r]);
      vhb[2][m][col] = f2bf(a2[r]);
      catA[m][128 + col] = f2bf(sqrtf(fmaxf(n2[r], 1e-8f)));
    }
  }
  __syncthreads();
  // GVP0 s (K=256, N=512: 4 tiles/wave) + wv0 gate, in regs
  {
    v8f accS[4];
#pragma unroll
    for (int q = 0; q < 4; ++q)
      accS[q] = gemm_tile(&catA[0][0], 640, w.ws0, (wave + 8 * q) * 16, 256, lane);
    v8f a0 = gemm_tile(&vhb[0][0][0], 128, w.wv0, wave * 16, 128, lane);
    v8f a1 = gemm_tile(&vhb[1][0][0], 128, w.wv0, wave * 16, 128, lane);
    v8f a2 = gemm_tile(&vhb[2][0][0], 128, w.wv0, wave * 16, 128, lane);
    __syncthreads();
    const int mb = (lane & 16) ? 8 : 0;
#pragma unroll
    for (int q = 0; q < 4; ++q) {
      const int col = (wave + 8 * q) * 16 + (lane & 15);
#pragma unroll
      for (int r = 0; r < 8; ++r)
        catA[mb + r][col] = f2bf(fmaxf(accS[q][r] + w.b0[col], 0.f));
    }
    v8f n2 = a0 * a0 + a1 * a1 + a2 * a2;
    const int col2 = wave * 16 + (lane & 15);
#pragma unroll
    for (int r = 0; r < 8; ++r) {
      const int m = mb + r;
      const float gt = sigm(sqrtf(fmaxf(n2[r], 1e-8f)));
      vbf[0][m][col2] = f2bf(a0[r] * gt);
      vbf[1][m][col2] = f2bf(a1[r] * gt);
      vbf[2][m][col2] = f2bf(a2[r] * gt);
    }
  }
  __syncthreads();
  // GVP1 vh (K=128, N=128); vn1 -> catA[:,512..640)
  {
    const int t = wave;
    v8f a0 = gemm_tile(&vbf[0][0][0], 128, w.wh1, t * 16, 128, lane);
    v8f a1 = gemm_tile(&vbf[1][0][0], 128, w.wh1, t * 16, 128, lane);
    v8f a2 = gemm_tile(&vbf[2][0][0], 128, w.wh1, t * 16, 128, lane);
    v8f n2 = a0 * a0 + a1 * a1 + a2 * a2;
    const int col = t * 16 + (lane & 15), mb = (lane & 16) ? 8 : 0;
#pragma unroll
    for (int r = 0; r < 8; ++r) {
      const int m = mb + r;
      vhb[0][m][col] = f2bf(a0[r]); vhb[1][m][col] = f2bf(a1[r]);
      vhb[2][m][col] = f2bf(a2[r]);
      catA[m][512 + col] = f2bf(sqrtf(fmaxf(n2[r], 1e-8f)));
    }
  }
  __syncthreads();
  // GVP1 outputs (no act): plain store of fs / fv
  {
    v8f aS = gemm_tile(&catA[0][0], 640, w.ws1, wave * 16, 640, lane);
    const int col = wave * 16 + (lane & 15), mb = (lane & 16) ? 8 : 0;
#pragma unroll
    for (int r = 0; r < 8; ++r)
      fs[(n0 + mb + r) * 128 + col] = aS[r] + w.b1[col];
  }
  for (int t = wave; t < 4; t += 8) {
    v8f a0 = gemm_tile(&vhb[0][0][0], 128, w.wv1, t * 16, 128, lane);
    v8f a1 = gemm_tile(&vhb[1][0][0], 128, w.wv1, t * 16, 128, lane);
    v8f a2 = gemm_tile(&vhb[2][0][0], 128, w.wv1, t * 16, 128, lane);
    const int o = t * 16 + (lane & 15), mb = (lane & 16) ? 8 : 0;
#pragma unroll
    for (int r = 0; r < 8; ++r) {
      const int node = n0 + mb + r;
      fv[node * 192 + o * 3 + 0] = a0[r];
      fv[node * 192 + o * 3 + 1] = a1[r];
      fv[node * 192 + o * 3 + 2] = a2[r];
    }
  }
}

// ---------------- residual + GVP LayerNorm (wave per node) ------------------
__global__ __launch_bounds__(256) void residual_ln(
    const float* __restrict__ s_in, const float* __restrict__ v_in,
    const float* __restrict__ dsb, const float* __restrict__ dvb,
    const float* __restrict__ deg, const float* __restrict__ g,
    const float* __restrict__ b, float* __restrict__ s_out,
    float* __restrict__ v_out, int Nn) {
  const int lane = threadIdx.x & 31;
  const int node = blockIdx.x * 8 + (threadIdx.x >> 5);
  if (node >= Nn) return;
  float invd = 1.f;
  if (deg) invd = 1.f / fmaxf(deg[node], 1.f);
  float x[4]; float sum = 0.f;
#pragma unroll
  for (int t = 0; t < 4; ++t) {
    const int k = lane + 32 * t;
    float val = s_in[node * 128 + k];
    if (dsb) val += dsb[node * 128 + k] * invd;
    x[t] = val; sum += val;
  }
  const float mu = wred(sum) * (1.f / 128.f);
  float vs = 0.f;
#pragma unroll
  for (int t = 0; t < 4; ++t) { const float d = x[t] - mu; vs += d * d; }
  const float inv = rsqrtf(wred(vs) * (1.f / 128.f) + 1e-5f);
#pragma unroll
  for (int t = 0; t < 4; ++t) {
    const int k = lane + 32 * t;
    s_out[node * 128 + k] = (x[t] - mu) * inv * g[k] + b[k];
  }
  float vv[6]; float ss = 0.f;
#pragma unroll
  for (int p = 0; p < 2; ++p) {
    const int o = lane * 2 + p;
    float a = v_in[node * 192 + o * 3 + 0];
    float bb = v_in[node * 192 + o * 3 + 1];
    float cc = v_in[node * 192 + o * 3 + 2];
    if (dvb) {
      a  += dvb[node * 192 + o * 3 + 0] * invd;
      bb += dvb[node * 192 + o * 3 + 1] * invd;
      cc += dvb[node * 192 + o * 3 + 2] * invd;
    }
    vv[p * 3 + 0] = a; vv[p * 3 + 1] = bb; vv[p * 3 + 2] = cc;
    ss += fmaxf(a * a + bb * bb + cc * cc, 1e-8f);
  }
  const float scale = rsqrtf(wred(ss) * (1.f / 64.f));
#pragma unroll
  for (int p = 0; p < 2; ++p) {
    const int o = lane * 2 + p;
    v_out[node * 192 + o * 3 + 0] = vv[p * 3 + 0] * scale;
    v_out[node * 192 + o * 3 + 1] = vv[p * 3 + 1] * scale;
    v_out[node * 192 + o * 3 + 2] = vv[p * 3 + 2] * scale;
  }
}

// ---------------- output GVP + global_add_pool ------------------------------
__global__ __launch_bounds__(256) void out_gvp(
    const float* __restrict__ s, const float* __restrict__ v,
    const int* __restrict__ batch, const unsigned short* __restrict__ whP,
    const unsigned short* __restrict__ wsP, const float* __restrict__ wsb,
    float* __restrict__ out) {
  __shared__ alignas(16) unsigned short catA[16][192];
  __shared__ alignas(16) unsigned short vbf[3][16][64];
  __shared__ int bt[16];
  const int tid = threadIdx.x, lane = tid & 31, wave = tid >> 5;
  const int n0 = blockIdx.x * 16;
  for (int i = tid; i < 16 * 128; i += 256) {
    const int m = i >> 7, k = i & 127;
    catA[m][k] = f2bf(s[(n0 + m) * 128 + k]);
  }
  for (int i = tid; i < 16 * 64; i += 256) {
    const int m = i >> 6, o = i & 63;
#pragma unroll
    for (int c = 0; c < 3; ++c) vbf[c][m][o] = f2bf(v[(n0 + m) * 192 + o * 3 + c]);
  }
  if (tid < 16) bt[tid] = batch[n0 + tid];
  __syncthreads();
  for (int t = wave; t < 4; t += 8) {  // wh: K=64, N=64 -> vn only (vo=0)
    v8f a0 = gemm_tile(&vbf[0][0][0], 64, whP, t * 16, 64, lane);
    v8f a1 = gemm_tile(&vbf[1][0][0], 64, whP, t * 16, 64, lane);
    v8f a2 = gemm_tile(&vbf[2][0][0], 64, whP, t * 16, 64, lane);
    v8f n2 = a0 * a0 + a1 * a1 + a2 * a2;
    const int col = t * 16 + (lane & 15), mb = (lane & 16) ? 8 : 0;
#pragma unroll
    for (int r = 0; r < 8; ++r)
      catA[mb + r][128 + col] = f2bf(sqrtf(fmaxf(n2[r], 1e-8f)));
  }
  __syncthreads();
  {
    v8f aS = gemm_tile(&catA[0][0], 192, wsP, wave * 16, 192, lane);
    const int col = wave * 16 + (lane & 15), mb = (lane & 16) ? 8 : 0;
#pragma unroll
    for (int r = 0; r < 8; ++r)
      atomicAdd(&out[bt[mb + r] * 128 + col], fmaxf(aS[r] + wsb[col], 0.f));
  }
}

// ======================= host orchestration =================================
extern "C" void kernel_launch(void* const* d_in, const int* in_sizes, int n_in,
                              void* d_out, int out_size, void* d_ws, size_t ws_size,
                              hipStream_t stream) {
  // Input order: setup_inputs() dict insertion order, params flattened
  // depth-first (dict insertion order; lists by index; gvp = wh,ws_w,ws_b[,wv]).
  int ii = 0;
  auto F = [&]() { return (const float*)d_in[ii++]; };
  const float *Wv_g = F(), *Wv_b = F();
  const float *Wv_wh = F(), *Wv_ws = F(), *Wv_wsb = F(), *Wv_wv = F();
  const float *We_g = F(), *We_b = F();
  const float *We_wh = F(), *We_ws = F(), *We_wsb = F(), *We_wv = F();
  struct LayerP {
    const float *m_wh[3], *m_ws[3], *m_wsb[3], *m_wv[3];
    const float *n0g, *n0b, *n1g, *n1b;
    const float *f_wh[2], *f_ws[2], *f_wsb[2], *f_wv[2];
  } L[3];
  for (int l = 0; l < 3; ++l) {
    for (int k = 0; k < 3; ++k) {
      L[l].m_wh[k] = F(); L[l].m_ws[k] = F(); L[l].m_wsb[k] = F(); L[l].m_wv[k] = F();
    }
    L[l].n0g = F(); L[l].n0b = F();
    L[l].n1g = F(); L[l].n1b = F();
    for (int k = 0; k < 2; ++k) {
      L[l].f_wh[k] = F(); L[l].f_ws[k] = F(); L[l].f_wsb[k] = F(); L[l].f_wv[k] = F();
    }
  }
  const float *Wo_g = F(), *Wo_b = F();
  const float *Wo_wh = F(), *Wo_ws = F(), *Wo_wsb = F();
  const int idx_ns = ii;  const float* node_s = F();
  const float* node_v = F();
  const int idx_es = ii;  const float* edge_s = F();
  const float* edge_v = F();
  const int* edge_index = (const int*)d_in[ii++];
  const int* batch = (const int*)d_in[ii++];
  (void)n_in;
  const int Nn = in_sizes[idx_ns] / 66;
  const int E  = in_sizes[idx_es] / 16;

  // workspace bump allocator (256B aligned)
  size_t off = 0;
  auto alloc = [&](size_t bytes) -> void* {
    void* p = (char*)d_ws + off;
    off += (bytes + 255) & ~(size_t)255;
    return p;
  };
  float* s   = (float*)alloc((size_t)Nn * 128 * 4);
  float* v   = (float*)alloc((size_t)Nn * 192 * 4);
  float* esb = (float*)alloc((size_t)E * 32 * 4);
  float* evb = (float*)alloc((size_t)E * 3 * 4);
  float* deg = (float*)alloc((size_t)Nn * 4);
  float* dsb = (float*)alloc((size_t)Nn * 128 * 4);
  float* dvb = (float*)alloc((size_t)Nn * 192 * 4);
  (void)ws_size;

  auto pack = [&](const float* W, int K, int N) -> const unsigned short* {
    const int Kp = (K + 31) & ~31, Np = (N + 15) & ~15;
    unsigned short* o = (unsigned short*)alloc((size_t)Kp * Np * 2);
    const int tot = Kp * Np;
    pack_w<<<(tot + 255) / 256, 256, 0, stream>>>(W, o, K, N, Kp, Np);
    return o;
  };

  const unsigned short* WvP_ws = pack(Wv_ws, 130, 128);
  const unsigned short* WvP_wv = pack(Wv_wv, 64, 64);
  MsgW mw[3]; FFW fw[3];
  for (int l = 0; l < 3; ++l) {
    mw[l].wh0 = pack(L[l].m_wh[0], 129, 129);
    mw[l].ws0 = pack(L[l].m_ws[0], 417, 128);
    mw[l].wv0 = pack(L[l].m_wv[0], 129, 64);
    mw[l].wh1 = pack(L[l].m_wh[1], 64, 64);
    mw[l].ws1 = pack(L[l].m_ws[1], 192, 128);
    mw[l].wv1 = pack(L[l].m_wv[1], 64, 64);
    mw[l].wh2 = pack(L[l].m_wh[2], 64, 64);
    mw[l].ws2 = pack(L[l].m_ws[2], 192, 128);
    mw[l].wv2 = pack(L[l].m_wv[2], 64, 64);
    mw[l].b0 = L[l].m_wsb[0]; mw[l].b1 = L[l].m_wsb[1]; mw[l].b2 = L[l].m_wsb[2];
    fw[l].wh0 = pack(L[l].f_wh[0], 64, 128);
    fw[l].ws0 = pack(L[l].f_ws[0], 256, 512);
    fw[l].wv0 = pack(L[l].f_wv[0], 128, 128);
    fw[l].wh1 = pack(L[l].f_wh[1], 128, 128);
    fw[l].ws1 = pack(L[l].f_ws[1], 640, 128);
    fw[l].wv1 = pack(L[l].f_wv[1], 128, 64);
    fw[l].b0 = L[l].f_wsb[0]; fw[l].b1 = L[l].f_wsb[1];
  }
  const unsigned short* WoP_wh = pack(Wo_wh, 64, 64);
  const unsigned short* WoP_ws = pack(Wo_ws, 192, 128);

  hipMemsetAsync(deg, 0, (size_t)Nn * 4, stream);
  deg_kernel<<<(E + 255) / 256, 256, 0, stream>>>(edge_index, deg, E);
  node_embed<<<Nn / 16, 256, 0, stream>>>(node_s, node_v, Wv_g, Wv_b, Wv_wh,
                                          WvP_ws, Wv_wsb, WvP_wv, s, v);
  edge_embed<<<(E + 255) / 256, 256, 0, stream>>>(edge_s, edge_v, We_g, We_b,
                                                  We_wh, We_ws, We_wsb, We_wv,
                                                  esb, evb, E);
  for (int l = 0; l < 3; ++l) {
    hipMemsetAsync(dsb, 0, (size_t)Nn * 128 * 4, stream);
    hipMemsetAsync(dvb, 0, (size_t)Nn * 192 * 4, stream);
    msg_fused<<<E / 16, 256, 0, stream>>>(s, v, esb, evb, edge_index, mw[l],
                                          dsb, dvb, E);
    residual_ln<<<(Nn + 7) / 8, 256, 0, stream>>>(s, v, dsb, dvb, deg,
                                                  L[l].n0g, L[l].n0b, s, v, Nn);
    ff_fused<<<Nn / 16, 256, 0, stream>>>(s, v, fw[l], dsb, dvb);
    residual_ln<<<(Nn + 7) / 8, 256, 0, stream>>>(s, v, dsb, dvb, nullptr,
                                                  L[l].n1g, L[l].n1b, s, v, Nn);
  }
  // final LN (into dsb/dvb), output GVP + pooled atomics
  residual_ln<<<(Nn + 7) / 8, 256, 0, stream>>>(s, v, nullptr, nullptr, nullptr,
                                                Wo_g, Wo_b, dsb, dvb, Nn);
  hipMemsetAsync(d_out, 0, (size_t)out_size * 4, stream);
  out_gvp<<<Nn / 16, 256, 0, stream>>>(dsb, dvb, batch, WoP_wh, WoP_ws, Wo_wsb,
                                       (float*)d_out);
}
